// ProHealth_VAE_42107859370000
// MI455X (gfx1250) — matrix operand order repaired and checked by reading, not compiled, not run on hardware
//
#include <hip/hip_runtime.h>
#include <hip/hip_bf16.h>

// ---------------------------------------------------------------------------
// ProHealth VAE / GRU-ODE pipeline for MI455X (gfx1250), bf16 WMMA everywhere.
// B=256 T=64 C=4880 D=128 H=1024 NT=101 ND=10
// ---------------------------------------------------------------------------

constexpr int kB  = 256;
constexpr int kT  = 64;
constexpr int kC  = 4880;
constexpr int kD  = 128;
constexpr int kH  = 1024;
constexpr int kNT = 101;
constexpr int kND = 10;

typedef __attribute__((ext_vector_type(16))) __bf16 v16bf;
typedef __attribute__((ext_vector_type(8)))  float  v8f;

__device__ __forceinline__ v8f wmma_bf16(v16bf a, v16bf b, v8f c) {
  // D = A(16x32 bf16) * B(32x16 bf16) + C(16x16 f32)
  return __builtin_amdgcn_wmma_f32_16x16x32_bf16(
      /*neg_a=*/false, a, /*neg_b=*/false, b,
      /*c_mod=*/(short)0, c, /*reuse_a=*/false, /*reuse_b=*/false);
}

// A-fragment: p = &A[(m0 + (lane&15)) * lda + k0], koff = (lane>>4)*8
// lane<16 holds K in {0..7, 16..23}; lane>=16 holds K in {8..15, 24..31}
__device__ __forceinline__ v16bf load_a(const __bf16* p, int koff) {
  v16bf a;
#pragma unroll
  for (int i = 0; i < 8; ++i) a[i] = p[koff + i];
#pragma unroll
  for (int i = 0; i < 8; ++i) a[8 + i] = p[16 + koff + i];
  return a;
}

// B-fragment (B[k,n] = W[j0+n, k0+k]): p = &W[(j0 + (lane&15))*ldw + k0 + (lane>>4)*8]
// lanes 0-15 hold K in {0..7, 16..23} of column (lane&15); lanes 16-31 hold K in {8..15, 24..31}.
__device__ __forceinline__ v16bf load_b(const __bf16* p) {
  v16bf b;
#pragma unroll
  for (int i = 0; i < 8; ++i) b[i] = p[i];
#pragma unroll
  for (int i = 0; i < 8; ++i) b[8 + i] = p[16 + i];
  return b;
}

__device__ __forceinline__ float sigmoidf_(float x) { return 1.0f / (1.0f + __expf(-x)); }

// ---------------------------------------------------------------------------
// Prep kernels
// ---------------------------------------------------------------------------
__global__ __launch_bounds__(256) void f32_to_bf16_kernel(const float* __restrict__ src,
                                                          __bf16* __restrict__ dst, int n) {
  int i = blockIdx.x * 256 + threadIdx.x;
  if (i < n) dst[i] = (__bf16)src[i];
}

__global__ __launch_bounds__(256) void zero_f32_kernel(float* __restrict__ p, int n) {
  int i = blockIdx.x * 256 + threadIdx.x;
  if (i < n) p[i] = 0.0f;
}

// Sparse visit embedding: ve[b,t,:] = sum_c Hs[b,c,t] * emb[c,:]  (bf16 out)
__global__ __launch_bounds__(128) void embed_kernel(const float* __restrict__ Hs,
                                                    const float* __restrict__ emb,
                                                    __bf16* __restrict__ ve) {
  const int bt = blockIdx.x;          // b*T + t
  const int b = bt / kT, t = bt % kT;
  const int d = threadIdx.x;          // 0..127
  __shared__ float wcache[128];
  const float* hcol = Hs + (size_t)b * kC * kT + t;   // stride kT per code c
  float acc = 0.0f;
  for (int c0 = 0; c0 < kC; c0 += 128) {
    int c = c0 + threadIdx.x;
    wcache[threadIdx.x] = (c < kC) ? hcol[(size_t)c * kT] : 0.0f;
    __syncthreads();
    int lim = min(128, kC - c0);
    for (int i = 0; i < lim; ++i) {
      float w = wcache[i];
      if (w != 0.0f) acc += w * emb[(size_t)(c0 + i) * kD + d];
    }
    __syncthreads();
  }
  ve[(size_t)bt * kD + d] = (__bf16)acc;
}

// ---------------------------------------------------------------------------
// GRU step: fused [256x1024]x[1024x3x1024] (Whh) + [256x128]x[128x3x1024] (Wih)
// Block = 8 waves (4x2), block tile 64(M) x 32(N); grid (32, 4).
// ---------------------------------------------------------------------------
__global__ __launch_bounds__(256) void gru_step_kernel(
    const __bf16* __restrict__ h_bf, const float* __restrict__ h_f32,
    const __bf16* __restrict__ ve_bf,
    const __bf16* __restrict__ Whh_bf, const __bf16* __restrict__ Wih_bf,
    const float* __restrict__ bih, const float* __restrict__ bhh,
    float* __restrict__ hout_f32, __bf16* __restrict__ hout_bf, int t) {
  const int lane = threadIdx.x & 31;
  const int wid  = threadIdx.x >> 5;
  const int m0 = blockIdx.y * 64 + (wid >> 1) * 16;
  const int j0 = blockIdx.x * 32 + (wid & 1) * 16;
  const int arow = lane & 15, koff = (lane >> 4) * 8;
  const int bcol = lane & 15, kbase = (lane >> 4) * 8;

  v8f acc_r = {}, acc_z = {}, acc_in = {}, acc_hn = {};

  const __bf16* Ap = h_bf + (size_t)(m0 + arow) * kH;
  const __bf16* Br = Whh_bf + (size_t)(j0 + bcol) * kH + kbase;
  const __bf16* Bz = Whh_bf + (size_t)(kH + j0 + bcol) * kH + kbase;
  const __bf16* Bn = Whh_bf + (size_t)(2 * kH + j0 + bcol) * kH + kbase;
#pragma unroll 4
  for (int k0 = 0; k0 < kH; k0 += 32) {
    if (k0 + 64 < kH) __builtin_prefetch(Ap + k0 + 64, 0, 1);
    v16bf a = load_a(Ap + k0, koff);
    acc_r  = wmma_bf16(a, load_b(Br + k0), acc_r);
    acc_z  = wmma_bf16(a, load_b(Bz + k0), acc_z);
    acc_hn = wmma_bf16(a, load_b(Bn + k0), acc_hn);
  }

  const __bf16* Av = ve_bf + ((size_t)(m0 + arow) * kT + t) * kD;
  const __bf16* Cr = Wih_bf + (size_t)(j0 + bcol) * kD + kbase;
  const __bf16* Cz = Wih_bf + (size_t)(kH + j0 + bcol) * kD + kbase;
  const __bf16* Cn = Wih_bf + (size_t)(2 * kH + j0 + bcol) * kD + kbase;
#pragma unroll
  for (int k0 = 0; k0 < kD; k0 += 32) {
    v16bf a = load_a(Av + k0, koff);
    acc_r  = wmma_bf16(a, load_b(Cr + k0), acc_r);
    acc_z  = wmma_bf16(a, load_b(Cz + k0), acc_z);
    acc_in = wmma_bf16(a, load_b(Cn + k0), acc_in);
  }

#pragma unroll
  for (int g = 0; g < 8; ++g) {
    int row = m0 + (lane >> 4) * 8 + g;
    int col = j0 + (lane & 15);
    float r = sigmoidf_(acc_r[g] + bih[col] + bhh[col]);
    float z = sigmoidf_(acc_z[g] + bih[kH + col] + bhh[kH + col]);
    float n = tanhf(acc_in[g] + bih[2 * kH + col] + r * (acc_hn[g] + bhh[2 * kH + col]));
    float ho = h_f32[(size_t)row * kH + col];
    float hn = (1.0f - z) * n + z * ho;
    hout_f32[(size_t)row * kH + col] = hn;
    hout_bf[(size_t)row * kH + col] = (__bf16)hn;
  }
}

// Streaming softmax attention accumulate: num += e^{h·w} h ; den += e^{h·w}
__global__ __launch_bounds__(256) void attn_accum_kernel(const float* __restrict__ hstep,
                                                         const float* __restrict__ attn_w,
                                                         float* __restrict__ num,
                                                         float* __restrict__ den) {
  const int b = blockIdx.x, tid = threadIdx.x;
  __shared__ float red[256];
  float s = 0.0f;
  for (int i = tid; i < kH; i += 256) s += hstep[(size_t)b * kH + i] * attn_w[i];
  red[tid] = s;
  __syncthreads();
  for (int off = 128; off > 0; off >>= 1) {
    if (tid < off) red[tid] += red[tid + off];
    __syncthreads();
  }
  float e = __expf(red[0]);
  for (int i = tid; i < kH; i += 256) num[(size_t)b * kH + i] += e * hstep[(size_t)b * kH + i];
  if (tid == 0) den[b] += e;
}

__global__ __launch_bounds__(256) void henc_kernel(const float* __restrict__ num,
                                                   const float* __restrict__ den,
                                                   __bf16* __restrict__ henc_bf) {
  int i = blockIdx.x * 256 + threadIdx.x;  // B*H
  if (i < kB * kH) henc_bf[i] = (__bf16)(num[i] / den[i / kH]);
}

// ---------------------------------------------------------------------------
// VAE head: mu/logvar GEMMs + reparameterization; emits pred_z[0] rows too.
// ---------------------------------------------------------------------------
__global__ __launch_bounds__(256) void vae_kernel(
    const __bf16* __restrict__ henc_bf, const __bf16* __restrict__ Wmu,
    const __bf16* __restrict__ Wvar, const float* __restrict__ bmu,
    const float* __restrict__ bvar, const float* __restrict__ eps,
    const int* __restrict__ last_idx, float* __restrict__ z_f32,
    __bf16* __restrict__ z_bf, float* __restrict__ out_last) {
  const int lane = threadIdx.x & 31;
  const int wid  = threadIdx.x >> 5;
  const int m0 = blockIdx.y * 64 + (wid >> 1) * 16;
  const int j0 = blockIdx.x * 32 + (wid & 1) * 16;
  const int arow = lane & 15, koff = (lane >> 4) * 8;
  const int bcol = lane & 15, kbase = (lane >> 4) * 8;

  v8f acc_m = {}, acc_v = {};
  const __bf16* Ap = henc_bf + (size_t)(m0 + arow) * kH;
  const __bf16* Bm = Wmu + (size_t)(j0 + bcol) * kH + kbase;
  const __bf16* Bv = Wvar + (size_t)(j0 + bcol) * kH + kbase;
#pragma unroll 4
  for (int k0 = 0; k0 < kH; k0 += 32) {
    v16bf a = load_a(Ap + k0, koff);
    acc_m = wmma_bf16(a, load_b(Bm + k0), acc_m);
    acc_v = wmma_bf16(a, load_b(Bv + k0), acc_v);
  }
#pragma unroll
  for (int g = 0; g < 8; ++g) {
    int row = m0 + (lane >> 4) * 8 + g;
    int col = j0 + (lane & 15);
    float mu = acc_m[g] + bmu[col];
    float lv = acc_v[g] + bvar[col];
    float z0 = mu + __expf(0.5f * lv) * eps[(size_t)row * kH + col];
    z_f32[(size_t)row * kH + col] = z0;
    z_bf[(size_t)row * kH + col] = (__bf16)z0;
    if (last_idx[row] == 0) out_last[(size_t)row * kH + col] = z0;
  }
}

// ---------------------------------------------------------------------------
// GRU-ODE f(h): zz = sigmoid(h Whz^T); nn = tanh((zz*h) Whn^T); f = (1-zz)(nn-h)
// ---------------------------------------------------------------------------
__global__ __launch_bounds__(256) void ode_f1_kernel(
    const __bf16* __restrict__ hin_bf, const float* __restrict__ hin_f32,
    const __bf16* __restrict__ Whz, float* __restrict__ zz_f32,
    __bf16* __restrict__ g_bf) {
  const int lane = threadIdx.x & 31;
  const int wid  = threadIdx.x >> 5;
  const int m0 = blockIdx.y * 64 + (wid >> 1) * 16;
  const int j0 = blockIdx.x * 32 + (wid & 1) * 16;
  const int arow = lane & 15, koff = (lane >> 4) * 8;
  const int bcol = lane & 15, kbase = (lane >> 4) * 8;

  v8f acc = {};
  const __bf16* Ap = hin_bf + (size_t)(m0 + arow) * kH;
  const __bf16* Bp = Whz + (size_t)(j0 + bcol) * kH + kbase;
#pragma unroll 4
  for (int k0 = 0; k0 < kH; k0 += 32) {
    v16bf a = load_a(Ap + k0, koff);
    acc = wmma_bf16(a, load_b(Bp + k0), acc);
  }
#pragma unroll
  for (int g = 0; g < 8; ++g) {
    int row = m0 + (lane >> 4) * 8 + g;
    int col = j0 + (lane & 15);
    float zz = sigmoidf_(acc[g]);
    zz_f32[(size_t)row * kH + col] = zz;
    g_bf[(size_t)row * kH + col] = (__bf16)(zz * hin_f32[(size_t)row * kH + col]);
  }
}

// Second half of f + RK4 stage combine. stage: 0..3.
//  k = (1-zz)*(tanh(g Whn^T) - hin)
//  kacc = (stage==0) ? k : kacc + w*k  (w = 2,2,1 for stages 1,2,3)
//  stage<3 : zout = zbase + c*dt*k  (c = 0.5,0.5,1.0)
//  stage==3: zout = zbase + dt/6 * kacc ; capture pred_z[step+1] rows.
__global__ __launch_bounds__(256) void ode_f2_kernel(
    const __bf16* __restrict__ g_in_bf, const __bf16* __restrict__ Whn,
    const float* zz_f32, const float* hin_f32, const float* zbase_f32,
    float* kacc, float* zout_f32, __bf16* zout_bf,
    const float* __restrict__ ts, const int* __restrict__ last_idx,
    float* __restrict__ out_last, int step, int stage) {
  const int lane = threadIdx.x & 31;
  const int wid  = threadIdx.x >> 5;
  const int m0 = blockIdx.y * 64 + (wid >> 1) * 16;
  const int j0 = blockIdx.x * 32 + (wid & 1) * 16;
  const int arow = lane & 15, koff = (lane >> 4) * 8;
  const int bcol = lane & 15, kbase = (lane >> 4) * 8;

  v8f acc = {};
  const __bf16* Ap = g_in_bf + (size_t)(m0 + arow) * kH;
  const __bf16* Bp = Whn + (size_t)(j0 + bcol) * kH + kbase;
#pragma unroll 4
  for (int k0 = 0; k0 < kH; k0 += 32) {
    v16bf a = load_a(Ap + k0, koff);
    acc = wmma_bf16(a, load_b(Bp + k0), acc);
  }
  const float dt = ts[step + 1] - ts[step];
#pragma unroll
  for (int g = 0; g < 8; ++g) {
    int row = m0 + (lane >> 4) * 8 + g;
    int col = j0 + (lane & 15);
    size_t idx = (size_t)row * kH + col;
    float zz = zz_f32[idx];
    float nn = tanhf(acc[g]);
    float k = (1.0f - zz) * (nn - hin_f32[idx]);
    float ka;
    if (stage == 0)      ka = k;
    else if (stage == 3) ka = kacc[idx] + k;
    else                 ka = kacc[idx] + 2.0f * k;
    kacc[idx] = ka;
    if (stage < 3) {
      float c = (stage == 2) ? 1.0f : 0.5f;
      float hnew = zbase_f32[idx] + c * dt * k;
      zout_f32[idx] = hnew;
      zout_bf[idx] = (__bf16)hnew;
    } else {
      float hnew = zbase_f32[idx] + (dt * (1.0f / 6.0f)) * ka;
      zout_f32[idx] = hnew;
      zout_bf[idx] = (__bf16)hnew;
      if (last_idx[row] == step + 1) out_last[idx] = hnew;
    }
  }
}

// LLM diagnosis mask scatter
__global__ __launch_bounds__(256) void mask_scatter_kernel(const int* __restrict__ llm_idx,
                                                           float* __restrict__ mask) {
  int i = blockIdx.x * 256 + threadIdx.x;
  if (i < kB * kND) {
    int b = i / kND;
    int c = llm_idx[i];
    mask[(size_t)b * kC + c] = 1.0f;
  }
}

// ---------------------------------------------------------------------------
// Host orchestration
// ---------------------------------------------------------------------------
extern "C" void kernel_launch(void* const* d_in, const int* in_sizes, int n_in,
                              void* d_out, int out_size, void* d_ws, size_t ws_size,
                              hipStream_t stream) {
  (void)in_sizes; (void)n_in; (void)out_size; (void)ws_size;

  const float* Hs        = (const float*)d_in[0];
  const float* ts        = (const float*)d_in[1];
  const int*   last_idx  = (const int*)d_in[2];
  const int*   llm_idx   = (const int*)d_in[3];
  const float* eps       = (const float*)d_in[4];
  const float* emb       = (const float*)d_in[5];
  const float* gru_Wih   = (const float*)d_in[6];
  const float* gru_Whh   = (const float*)d_in[7];
  const float* gru_bih   = (const float*)d_in[8];
  const float* gru_bhh   = (const float*)d_in[9];
  const float* attn_w    = (const float*)d_in[10];
  const float* fc_mu_W   = (const float*)d_in[11];
  const float* fc_mu_b   = (const float*)d_in[12];
  const float* fc_var_W  = (const float*)d_in[13];
  const float* fc_var_b  = (const float*)d_in[14];
  const float* ode_Whz   = (const float*)d_in[15];
  const float* ode_Whn   = (const float*)d_in[16];

  float* out_last = (float*)d_out;                       // [B,H]
  float* out_mask = (float*)d_out + (size_t)kB * kH;     // [B,C]

  // Workspace carve-out (256B aligned).
  char* w = (char*)d_ws;
  size_t off = 0;
  auto alloc = [&](size_t bytes) -> void* {
    void* p = w + off;
    off = (off + bytes + 255) & ~(size_t)255;
    return p;
  };
  __bf16* ve_bf   = (__bf16*)alloc((size_t)kB * kT * kD * 2);
  __bf16* Whh_bf  = (__bf16*)alloc((size_t)3 * kH * kH * 2);
  __bf16* Wih_bf  = (__bf16*)alloc((size_t)3 * kH * kD * 2);
  __bf16* Wmu_bf  = (__bf16*)alloc((size_t)kH * kH * 2);
  __bf16* Wvar_bf = (__bf16*)alloc((size_t)kH * kH * 2);
  __bf16* Whz_bf  = (__bf16*)alloc((size_t)kH * kH * 2);
  __bf16* Whn_bf  = (__bf16*)alloc((size_t)kH * kH * 2);
  float*  hA_f32  = (float*)alloc((size_t)kB * kH * 4);
  float*  hB_f32  = (float*)alloc((size_t)kB * kH * 4);
  __bf16* hA_bf   = (__bf16*)alloc((size_t)kB * kH * 2);
  __bf16* hB_bf   = (__bf16*)alloc((size_t)kB * kH * 2);
  float*  num     = (float*)alloc((size_t)kB * kH * 4);
  float*  den     = (float*)alloc((size_t)kB * 4);
  __bf16* henc_bf = (__bf16*)alloc((size_t)kB * kH * 2);
  float*  zb_f32  = (float*)alloc((size_t)kB * kH * 4);
  __bf16* zb_bf   = (__bf16*)alloc((size_t)kB * kH * 2);
  float*  st_f32  = (float*)alloc((size_t)kB * kH * 4);
  __bf16* st_bf   = (__bf16*)alloc((size_t)kB * kH * 2);
  float*  zz_f32  = (float*)alloc((size_t)kB * kH * 4);
  __bf16* g_bf    = (__bf16*)alloc((size_t)kB * kH * 2);
  float*  kacc    = (float*)alloc((size_t)kB * kH * 4);

  auto cvt = [&](const float* s, __bf16* d, int n) {
    f32_to_bf16_kernel<<<(n + 255) / 256, 256, 0, stream>>>(s, d, n);
  };
  auto zero = [&](float* p, int n) {
    zero_f32_kernel<<<(n + 255) / 256, 256, 0, stream>>>(p, n);
  };

  // --- prep: weight conversion + state init ---
  cvt(gru_Whh, Whh_bf, 3 * kH * kH);
  cvt(gru_Wih, Wih_bf, 3 * kH * kD);
  cvt(fc_mu_W, Wmu_bf, kH * kH);
  cvt(fc_var_W, Wvar_bf, kH * kH);
  cvt(ode_Whz, Whz_bf, kH * kH);
  cvt(ode_Whn, Whn_bf, kH * kH);
  zero(hA_f32, kB * kH);
  zero((float*)hA_bf, kB * kH / 2);  // bf16 zeros as raw words
  zero(num, kB * kH);
  zero(den, kB);

  // --- encoder: sparse visit embedding ---
  embed_kernel<<<kB * kT, 128, 0, stream>>>(Hs, emb, ve_bf);

  // --- GRU scan with streaming softmax attention ---
  const dim3 gemm_grid(kH / 32, kB / 64);  // 32 x 4
  for (int t = 0; t < kT; ++t) {
    const bool even = (t & 1) == 0;
    const float*  hin_f  = even ? hA_f32 : hB_f32;
    const __bf16* hin_b  = even ? hA_bf  : hB_bf;
    float*        hout_f = even ? hB_f32 : hA_f32;
    __bf16*       hout_b = even ? hB_bf  : hA_bf;
    gru_step_kernel<<<gemm_grid, 256, 0, stream>>>(
        hin_b, hin_f, ve_bf, Whh_bf, Wih_bf, gru_bih, gru_bhh, hout_f, hout_b, t);
    attn_accum_kernel<<<kB, 256, 0, stream>>>(hout_f, attn_w, num, den);
  }

  // --- henc + VAE reparameterization (also emits pred_z[0] rows) ---
  henc_kernel<<<(kB * kH + 255) / 256, 256, 0, stream>>>(num, den, henc_bf);
  vae_kernel<<<gemm_grid, 256, 0, stream>>>(henc_bf, Wmu_bf, Wvar_bf, fc_mu_b,
                                            fc_var_b, eps, last_idx, zb_f32, zb_bf,
                                            out_last);

  // --- GRU-ODE RK4 over NT-1 intervals ---
  for (int step = 0; step < kNT - 1; ++step) {
    for (int s = 0; s < 4; ++s) {
      const __bf16* hin_b = (s == 0) ? zb_bf : st_bf;
      const float*  hin_f = (s == 0) ? zb_f32 : st_f32;
      float*        zo_f  = (s == 3) ? zb_f32 : st_f32;
      __bf16*       zo_b  = (s == 3) ? zb_bf : st_bf;
      ode_f1_kernel<<<gemm_grid, 256, 0, stream>>>(hin_b, hin_f, Whz_bf, zz_f32, g_bf);
      ode_f2_kernel<<<gemm_grid, 256, 0, stream>>>(g_bf, Whn_bf, zz_f32, hin_f, zb_f32,
                                                   kacc, zo_f, zo_b, ts, last_idx,
                                                   out_last, step, s);
    }
  }

  // --- LLM diagnosis mask ---
  zero(out_mask, kB * kC);
  mask_scatter_kernel<<<(kB * kND + 255) / 256, 256, 0, stream>>>(llm_idx, out_mask);
}
